// RandomCausalMHA_9646496547174
// MI455X (gfx1250) — matrix operand order repaired and checked by reading, not compiled, not run on hardware
//
#include <hip/hip_runtime.h>
#include <hip/hip_bf16.h>

// ---------------------------------------------------------------------------
// Problem constants (B=4, T=2048, C=1024, H=16, D=64)
// ---------------------------------------------------------------------------
#define BB    4
#define TT    2048
#define CC    1024
#define HH    16
#define DD    64
#define C3    3072           // 3*C
#define MM    (BB * TT)      // 8192 rows in all GEMMs

typedef __attribute__((ext_vector_type(16))) __bf16        v16bf;
typedef __attribute__((ext_vector_type(8)))  float         v8f;
typedef __attribute__((ext_vector_type(8)))  unsigned int  v8u;
typedef __attribute__((ext_vector_type(4)))  unsigned int  v4u;

union BFV {
    v16bf v;
    struct { v4u lo; v4u hi; } u;
    v8u w;
};

__device__ __forceinline__ unsigned short f2bf(float f) {
    unsigned int u = __float_as_uint(f);
    u += 0x7fffu + ((u >> 16) & 1u);           // round-to-nearest-even
    return (unsigned short)(u >> 16);
}

// B-operand tile (32x16): two 16-byte chunks per lane (K 0..7 + K 16..23 etc).
__device__ __forceinline__ v16bf load_b16(const unsigned short* p) {
    BFV t; t.u.lo = *(const v4u*)p; t.u.hi = *(const v4u*)(p + 16); return t.v;
}
// A-operand tile (16x32): two 16-byte chunks per lane (K 0..7 + K 16..23 etc).
__device__ __forceinline__ v16bf load_a16(const unsigned short* p0,
                                          const unsigned short* p1) {
    BFV t; t.u.lo = *(const v4u*)p0; t.u.hi = *(const v4u*)p1; return t.v;
}

__device__ __forceinline__ v8f wmma_bf16(v16bf a, v16bf b, v8f c) {
    return __builtin_amdgcn_wmma_f32_16x16x32_bf16(
        /*neg_a=*/false, a, /*neg_b=*/false, b,
        /*c_mod=*/(short)0, c, /*reuse_a=*/false, /*reuse_b=*/false);
}

__device__ __forceinline__ float rowmax16(float x) {
#pragma unroll
    for (int off = 8; off >= 1; off >>= 1)
        x = fmaxf(x, __shfl_xor(x, off, 16));
    return x;
}
__device__ __forceinline__ float rowsum16(float x) {
#pragma unroll
    for (int off = 8; off >= 1; off >>= 1)
        x += __shfl_xor(x, off, 16);
    return x;
}

// ---------------------------------------------------------------------------
// fp32 -> bf16 conversion
// ---------------------------------------------------------------------------
__global__ void cvt_f32_bf16(const float* __restrict__ src,
                             unsigned short* __restrict__ dst, int n) {
    int i = blockIdx.x * blockDim.x + threadIdx.x;
    if (i < n) dst[i] = f2bf(src[i]);
}

// ---------------------------------------------------------------------------
// TN GEMM: Out[m,n] = sum_k A[m,k] * W[n,k]; one wave -> 32x64 output tile
// (2 M-tiles x 4 N-tiles; B operands reused across both A tiles ->
//  1.5 b128 loads per WMMA instead of 2.5). bf16 output.
// ---------------------------------------------------------------------------
__global__ __launch_bounds__(128)
void gemm_tn_bf16out(const unsigned short* __restrict__ A,
                     const unsigned short* __restrict__ W,
                     unsigned short* __restrict__ Out,
                     int tilesM32, int tilesN64, int N, int K) {
    const int lane = threadIdx.x & 31;
    const int wv   = threadIdx.x >> 5;
    const int gw   = blockIdx.x * 4 + wv;
    const int tm   = gw % tilesM32;
    const int tn   = gw / tilesM32;
    if (tn >= tilesN64) return;
    const int m0 = tm * 32, n0 = tn * 64;

    v8f acc[2][4] = {};
    const unsigned short* arow0 = A + (size_t)(m0 + (lane & 15)) * K + ((lane >> 4) * 8);
    const unsigned short* arow1 = arow0 + (size_t)16 * K;
    const unsigned short* brow[4];
#pragma unroll
    for (int t = 0; t < 4; ++t)
        brow[t] = W + (size_t)(n0 + t * 16 + (lane & 15)) * K + ((lane >> 4) * 8);

    for (int kk = 0; kk < K; kk += 32) {
        __builtin_prefetch(arow0 + kk + 128, 0, 1);   // global_prefetch_b8
        __builtin_prefetch(brow[0] + kk + 128, 0, 1);
        v16bf a0 = load_a16(arow0 + kk, arow0 + kk + 16);
        v16bf a1 = load_a16(arow1 + kk, arow1 + kk + 16);
#pragma unroll
        for (int t = 0; t < 4; ++t) {
            v16bf bt = load_b16(brow[t] + kk);
            acc[0][t] = wmma_bf16(a0, bt, acc[0][t]);
            acc[1][t] = wmma_bf16(a1, bt, acc[1][t]);
        }
    }

    const int rowOff = 8 * (lane >> 4);
#pragma unroll
    for (int h = 0; h < 2; ++h)
#pragma unroll
        for (int t = 0; t < 4; ++t)
#pragma unroll
            for (int r = 0; r < 8; ++r)
                Out[(size_t)(m0 + h * 16 + r + rowOff) * N + n0 + t * 16 + (lane & 15)] =
                    f2bf(acc[h][t][r]);
}

// Same GEMM but fp32 output + bias (final projection).
__global__ __launch_bounds__(128)
void gemm_tn_f32out(const unsigned short* __restrict__ A,
                    const unsigned short* __restrict__ W,
                    const float* __restrict__ bias,
                    float* __restrict__ Out,
                    int tilesM32, int tilesN64, int N, int K) {
    const int lane = threadIdx.x & 31;
    const int wv   = threadIdx.x >> 5;
    const int gw   = blockIdx.x * 4 + wv;
    const int tm   = gw % tilesM32;
    const int tn   = gw / tilesM32;
    if (tn >= tilesN64) return;
    const int m0 = tm * 32, n0 = tn * 64;

    v8f acc[2][4] = {};
    const unsigned short* arow0 = A + (size_t)(m0 + (lane & 15)) * K + ((lane >> 4) * 8);
    const unsigned short* arow1 = arow0 + (size_t)16 * K;
    const unsigned short* brow[4];
#pragma unroll
    for (int t = 0; t < 4; ++t)
        brow[t] = W + (size_t)(n0 + t * 16 + (lane & 15)) * K + ((lane >> 4) * 8);

    for (int kk = 0; kk < K; kk += 32) {
        __builtin_prefetch(arow0 + kk + 128, 0, 1);
        __builtin_prefetch(brow[0] + kk + 128, 0, 1);
        v16bf a0 = load_a16(arow0 + kk, arow0 + kk + 16);
        v16bf a1 = load_a16(arow1 + kk, arow1 + kk + 16);
#pragma unroll
        for (int t = 0; t < 4; ++t) {
            v16bf bt = load_b16(brow[t] + kk);
            acc[0][t] = wmma_bf16(a0, bt, acc[0][t]);
            acc[1][t] = wmma_bf16(a1, bt, acc[1][t]);
        }
    }

    const int rowOff = 8 * (lane >> 4);
#pragma unroll
    for (int t = 0; t < 4; ++t) {
        const float bn = bias[n0 + t * 16 + (lane & 15)];
#pragma unroll
        for (int h = 0; h < 2; ++h)
#pragma unroll
            for (int r = 0; r < 8; ++r)
                Out[(size_t)(m0 + h * 16 + r + rowOff) * N + n0 + t * 16 + (lane & 15)] =
                    acc[h][t][r] + bn;
    }
}

// ---------------------------------------------------------------------------
// Repack V to per-head transposed layout: vt[b][h][d][t] = qkv[b*T+t][2C + h*D + d]
// ---------------------------------------------------------------------------
__global__ void repack_vt(const unsigned short* __restrict__ qkv,
                          unsigned short* __restrict__ vt, int n) {
    int i = blockIdx.x * blockDim.x + threadIdx.x;
    if (i >= n) return;
    int t = i & (TT - 1);
    int rest = i >> 11;            // / T
    int d = rest & (DD - 1);
    rest >>= 6;                    // / D
    int h = rest & (HH - 1);
    int b = rest >> 4;             // / H
    vt[i] = qkv[(size_t)(b * TT + t) * C3 + 2 * CC + h * DD + d];
}

// ---------------------------------------------------------------------------
// Flash attention with permuted-causal mask. One wave = 16 queries of one
// (b, h). Keys processed in blocks of 32: 4 WMMA for Q·K^T (D=64), online
// softmax in f32, P staged bf16 through wave-private LDS (C-layout ->
// A-layout transpose), 4 WMMA for P·V. Output written bf16 to y.
// ---------------------------------------------------------------------------
#define NEG_BIG (-1.0e30f)
#define NEG_CHK (-1.0e29f)

__global__ __launch_bounds__(128)
void attn_flash(const unsigned short* __restrict__ qkv,
                const unsigned short* __restrict__ vt,
                const int* __restrict__ perm,
                unsigned short* __restrict__ y) {
    __shared__ unsigned short pbuf[4][16 * 32];

    const int lane = threadIdx.x & 31;
    const int wv   = threadIdx.x >> 5;
    const int gw   = blockIdx.x * 4 + wv;
    const int qt   = gw & (TT / 16 - 1);       // 128 query tiles
    const int rest = gw >> 7;
    const int h    = rest & (HH - 1);
    const int b    = rest >> 4;

    const int q0     = qt * 16;
    const int bT     = b * TT;
    const int hD     = h * DD;
    const int bhD    = (b * HH + h) * DD;
    const int rowOff = 8 * (lane >> 4);
    const int qiBase = q0 + rowOff;
    unsigned short* pw = pbuf[wv];

    // Q A-tiles: d 0..31 and d 32..63
    const unsigned short* qrow =
        qkv + (size_t)(bT + q0 + (lane & 15)) * C3 + hD + ((lane >> 4) * 8);
    const v16bf aq0 = load_a16(qrow,      qrow + 16);
    const v16bf aq1 = load_a16(qrow + 32, qrow + 48);

    v8f acc[4] = {};
    float m[8], l[8];
#pragma unroll
    for (int r = 0; r < 8; ++r) { m[r] = NEG_BIG; l[r] = 0.f; }

    const v8f zacc = {};
    const float SCALE = 0.125f;                // 1/sqrt(64)

    for (int kb = 0; kb < TT; kb += 32) {
        // K operand rows (d 0..7 + 16..23 per lane half)
        const unsigned short* kr0 =
            qkv + (size_t)(bT + kb + (lane & 15)) * C3 + CC + hD + ((lane >> 4) * 8);
        const unsigned short* kr1 = kr0 + (size_t)16 * C3;

        v8f s0 = wmma_bf16(aq0, load_b16(kr0), zacc);
        s0     = wmma_bf16(aq1, load_b16(kr0 + 32), s0);
        v8f s1 = wmma_bf16(aq0, load_b16(kr1), zacc);
        s1     = wmma_bf16(aq1, load_b16(kr1 + 32), s1);

        const int p0 = perm[kb + (lane & 15)];
        const int p1 = perm[kb + 16 + (lane & 15)];

        float pr0[8], pr1[8], corr[8];
#pragma unroll
        for (int r = 0; r < 8; ++r) {
            const int qi = qiBase + r;
            float v0 = s0[r] * SCALE; if (p0 > qi) v0 = NEG_BIG;
            float v1 = s1[r] * SCALE; if (p1 > qi) v1 = NEG_BIG;
            const float rm   = rowmax16(fmaxf(v0, v1));
            const float mnew = fmaxf(m[r], rm);
            const float c    = __expf(m[r] - mnew);
            const float e0   = (v0 <= NEG_CHK) ? 0.f : __expf(v0 - mnew);
            const float e1   = (v1 <= NEG_CHK) ? 0.f : __expf(v1 - mnew);
            l[r] = l[r] * c + rowsum16(e0 + e1);
            m[r] = mnew;
            corr[r] = c;
            pr0[r] = e0; pr1[r] = e1;
        }
#pragma unroll
        for (int t = 0; t < 4; ++t)
#pragma unroll
            for (int r = 0; r < 8; ++r)
                acc[t][r] *= corr[r];

        // C-layout -> LDS (16 rows x 32 keys, bf16)
#pragma unroll
        for (int r = 0; r < 8; ++r) {
            pw[(r + rowOff) * 32 + (lane & 15)]      = f2bf(pr0[r]);
            pw[(r + rowOff) * 32 + 16 + (lane & 15)] = f2bf(pr1[r]);
        }
        // LDS -> A-layout (16x32)
        const v16bf pa = load_a16(pw + (lane & 15) * 32 + ((lane >> 4) * 8),
                                  pw + (lane & 15) * 32 + 16 + ((lane >> 4) * 8));
        // P·V
#pragma unroll
        for (int t = 0; t < 4; ++t) {
            const unsigned short* vr =
                vt + (size_t)(bhD + t * 16 + (lane & 15)) * TT + kb + ((lane >> 4) * 8);
            acc[t] = wmma_bf16(pa, load_b16(vr), acc[t]);
        }
    }

    float linv[8];
#pragma unroll
    for (int r = 0; r < 8; ++r) linv[r] = 1.0f / l[r];
#pragma unroll
    for (int t = 0; t < 4; ++t)
#pragma unroll
        for (int r = 0; r < 8; ++r)
            y[(size_t)(bT + q0 + r + rowOff) * CC + hD + t * 16 + (lane & 15)] =
                f2bf(acc[t][r] * linv[r]);
}

// ---------------------------------------------------------------------------
// Launch
// ---------------------------------------------------------------------------
extern "C" void kernel_launch(void* const* d_in, const int* in_sizes, int n_in,
                              void* d_out, int out_size, void* d_ws, size_t ws_size,
                              hipStream_t stream) {
    const float* x      = (const float*)d_in[0];   // (B,T,C)
    const float* w_qkv  = (const float*)d_in[1];   // (3C,C)
    const float* w_proj = (const float*)d_in[2];   // (C,C)
    const float* b_proj = (const float*)d_in[3];   // (C,)
    const int*   perm   = (const int*)d_in[4];     // (T,)
    float*       out    = (float*)d_out;           // (B,T,C)

    // Workspace carve (all bf16 as ushort), 256B-aligned by construction.
    char* ws = (char*)d_ws;
    unsigned short* xb     = (unsigned short*)(ws + 0);          // 16 MB
    unsigned short* wqkvb  = (unsigned short*)(ws + 16777216);   //  6 MB
    unsigned short* wprojb = (unsigned short*)(ws + 23068672);   //  2 MB
    unsigned short* qkvb   = (unsigned short*)(ws + 25165824);   // 48 MB
    unsigned short* vtb    = (unsigned short*)(ws + 75497472);   // 16 MB
    unsigned short* yb     = (unsigned short*)(ws + 92274688);   // 16 MB

    const int nX = MM * CC;            // 8388608
    const int nWq = C3 * CC;           // 3145728
    const int nWp = CC * CC;           // 1048576
    const int nVt = BB * HH * DD * TT; // 8388608

    cvt_f32_bf16<<<(nX + 255) / 256, 256, 0, stream>>>(x, xb, nX);
    cvt_f32_bf16<<<(nWq + 255) / 256, 256, 0, stream>>>(w_qkv, wqkvb, nWq);
    cvt_f32_bf16<<<(nWp + 255) / 256, 256, 0, stream>>>(w_proj, wprojb, nWp);

    // QKV projection: M=8192, N=3072, K=1024 -> 256 x 48 wave tiles (32x64)
    {
        int tilesM = MM / 32, tilesN = C3 / 64;
        int blocks = (tilesM * tilesN) / 4;
        gemm_tn_bf16out<<<blocks, 128, 0, stream>>>(xb, wqkvb, qkvb,
                                                    tilesM, tilesN, C3, CC);
    }

    repack_vt<<<(nVt + 255) / 256, 256, 0, stream>>>(qkvb, vtb, nVt);

    // Attention: B*H*(T/16) = 8192 waves, 4 per block
    attn_flash<<<2048, 128, 0, stream>>>(qkvb, vtb, perm, yb);

    // Output projection: M=8192, N=1024, K=1024 -> 256 x 16 wave tiles (32x64)
    {
        int tilesM = MM / 32, tilesN = CC / 64;
        int blocks = (tilesM * tilesN) / 4;
        gemm_tn_f32out<<<blocks, 128, 0, stream>>>(yb, wprojb, b_proj, out,
                                                   tilesM, tilesN, CC, CC);
    }
}
